// WindowAttention_44521630990767
// MI455X (gfx1250) — matrix operand order repaired and checked by reading, not compiled, not run on hardware
//
#include <hip/hip_runtime.h>
#include <hip/hip_bf16.h>
#include <math.h>

// ---------------- Problem constants ----------------
#define BATCH 2
#define NC    6          // cameras
#define DIM   256
#define NHEAD 8
#define DH    32         // head dim
#define HW    50
#define QLEN  2500       // 50*50
#define QP    2512       // QLEN padded to mult of 16 (157 tiles)
#define K0    625        // pooled keys per cam
#define KP    640        // padded per-cam keys (40 tiles of 16)
#define ND    1536       // NC*DIM
#define SROW  3844       // LDS score row stride (floats), != mult of 64 banks

typedef __bf16 bf16_t;
typedef __attribute__((ext_vector_type(16))) __bf16 v16bf;
typedef __attribute__((ext_vector_type(8)))  float  v8f;

union FragU {
    uint4 u[2];
    unsigned short s[16];
    v16bf v;
};

__device__ __forceinline__ unsigned short f2bf(float x) {
    unsigned int u = __float_as_uint(x);
    unsigned int r = u + 0x7FFFu + ((u >> 16) & 1u);
    return (unsigned short)(r >> 16);
}

__device__ __forceinline__ v8f zero8() {
    v8f z;
#pragma unroll
    for (int i = 0; i < 8; ++i) z[i] = 0.0f;
    return z;
}

// ---------------- Pack kernels ----------------

// W[K][N] f32 -> Wt[N][K] bf16
__global__ void cvt_w_t(const float* __restrict__ Win, unsigned short* __restrict__ Wt,
                        int K, int N) {
    long t = (long)blockIdx.x * blockDim.x + threadIdx.x;
    if (t >= (long)K * N) return;
    int k = (int)(t / N);
    int n = (int)(t % N);
    Wt[(long)n * K + k] = f2bf(Win[t]);
}

// q (B,NC,DIM,50,50) f32 -> Xq (B,QP,NC*DIM) bf16 ; pad rows zero
__global__ void pack_q(const float* __restrict__ Qin, unsigned short* __restrict__ Xq) {
    long t = (long)blockIdx.x * blockDim.x + threadIdx.x;
    const long TOT = (long)BATCH * QP * ND;
    if (t >= TOT) return;
    int j  = (int)(t % ND);
    long r = t / ND;
    int qi = (int)(r % QP);
    int b  = (int)(r / QP);
    int n  = j / DIM, c = j % DIM;
    float v = 0.0f;
    if (qi < QLEN) v = Qin[(((long)(b * NC + n) * DIM + c) * QLEN) + qi];
    Xq[t] = f2bf(v);
}

// k/v (B,NC,DIM,50,50) f32 -> pooled 2x2 -> X (B,NC,KP,DIM) bf16 ; pad rows zero
__global__ void pack_pool(const float* __restrict__ In, unsigned short* __restrict__ Xo) {
    long t = (long)blockIdx.x * blockDim.x + threadIdx.x;
    const long TOT = (long)BATCH * NC * KP * DIM;
    if (t >= TOT) return;
    int c  = (int)(t % DIM);
    long r = t / DIM;
    int kp = (int)(r % KP);
    long r2 = r / KP;
    int n = (int)(r2 % NC);
    int b = (int)(r2 / NC);
    float v = 0.0f;
    if (kp < K0) {
        int i = kp / 25, j = kp % 25;
        long base = ((long)(b * NC + n) * DIM + c) * QLEN;
        int h0 = 2 * i, w0 = 2 * j;
        v = 0.25f * (In[base + h0 * HW + w0]     + In[base + h0 * HW + w0 + 1] +
                     In[base + (h0 + 1) * HW + w0] + In[base + (h0 + 1) * HW + w0 + 1]);
    }
    Xo[t] = f2bf(v);
}

// ---------------- Generic bf16 WMMA GEMM ----------------
// C[M,N] = A[M,K](bf16,row-major,lda) @ Wt[N,K](bf16) + bias
// Wave computes a 16x32 output (two N-tiles sharing one A fragment);
// block = 8 waves as 4(M) x 2(N pair) -> 64 x 64 tile.
// K-loop software pipelined with the last iteration peeled (no runtime guard).
// EP: 0=bf16 out (scaled by oscale), 1=f32 out, 2=bf16 out transposed (C[n*ldc+m]),
//     3=f32 out + residual R + dual bf16 store, 4=GELU(erf) bf16 out
template <int EP>
__global__ __launch_bounds__(256)
void gemm_bf16(const unsigned short* __restrict__ A, long sAz1, long sAz2, int lda,
               const unsigned short* __restrict__ Wt,
               const float* __restrict__ bias,
               void* __restrict__ Cout, long sCz1, long sCz2, int ldc,
               const float* __restrict__ R, long sRz1,
               unsigned short* __restrict__ C2, long sC2z1,
               int Mdim, int Ndim, int Kdim, int z2count, float oscale) {
    int z  = blockIdx.z;
    int z2 = z % z2count;
    int z1 = z / z2count;
    int wave = threadIdx.x >> 5;
    int lane = threadIdx.x & 31;
    int lo16 = lane & 15;
    int hi   = lane >> 4;
    int wm = wave >> 1, wn = wave & 1;           // 4 (M) x 2 (N-pair) waves
    int mBase = blockIdx.y * 64 + wm * 16;
    int nBase = blockIdx.x * 64 + wn * 32;       // 2 consecutive 16-col tiles
    if (mBase >= Mdim || nBase >= Ndim) return;

    const unsigned short* Ab = A + (long)z1 * sAz1 + (long)z2 * sAz2;
    const unsigned short* pa0 = Ab + (long)(mBase + lo16) * lda + (hi ? 8 : 0);
    const unsigned short* pb0 = Wt + (long)(nBase + lo16) * Kdim + (hi ? 8 : 0);
    const unsigned short* pb1 = pb0 + (long)16 * Kdim;

    v8f acc0 = zero8(), acc1 = zero8();

    FragU fa, fb0, fb1;
    fa.u[0]  = *(const uint4*)(pa0);
    fa.u[1]  = *(const uint4*)(pa0 + 16);
    fb0.u[0] = *(const uint4*)(pb0);
    fb0.u[1] = *(const uint4*)(pb0 + 16);
    fb1.u[0] = *(const uint4*)(pb1);
    fb1.u[1] = *(const uint4*)(pb1 + 16);

    int k0 = 0;
    for (; k0 + 32 < Kdim; k0 += 32) {           // steady state: prefetch unconditional
        FragU na, nb0, nb1;
        const unsigned short* qa  = pa0 + k0 + 32;
        const unsigned short* qb0 = pb0 + k0 + 32;
        const unsigned short* qb1 = pb1 + k0 + 32;
        na.u[0]  = *(const uint4*)(qa);
        na.u[1]  = *(const uint4*)(qa + 16);
        nb0.u[0] = *(const uint4*)(qb0);
        nb0.u[1] = *(const uint4*)(qb0 + 16);
        nb1.u[0] = *(const uint4*)(qb1);
        nb1.u[1] = *(const uint4*)(qb1 + 16);
        if (k0 + 64 < Kdim)                       // L2->near prefetch of streamed A
            __builtin_prefetch(pa0 + k0 + 64, 0, 1);
        acc0 = __builtin_amdgcn_wmma_f32_16x16x32_bf16(false, fa.v, false, fb0.v,
                                                       (short)0, acc0, false, false);
        acc1 = __builtin_amdgcn_wmma_f32_16x16x32_bf16(false, fa.v, false, fb1.v,
                                                       (short)0, acc1, false, false);
        fa = na; fb0 = nb0; fb1 = nb1;
    }
    // peeled final K-step
    acc0 = __builtin_amdgcn_wmma_f32_16x16x32_bf16(false, fa.v, false, fb0.v,
                                                   (short)0, acc0, false, false);
    acc1 = __builtin_amdgcn_wmma_f32_16x16x32_bf16(false, fa.v, false, fb1.v,
                                                   (short)0, acc1, false, false);

#pragma unroll
    for (int nt = 0; nt < 2; ++nt) {
        int ncol = nBase + nt * 16 + lo16;
        float bv = bias ? bias[ncol] : 0.0f;
        const v8f& acc = nt ? acc1 : acc0;
#pragma unroll
        for (int e = 0; e < 8; ++e) {
            int r = mBase + e + 8 * hi;
            float val = acc[e] + bv;
            if (EP == 0) {
                unsigned short* C = (unsigned short*)Cout + (long)z1 * sCz1 + (long)z2 * sCz2;
                C[(long)r * ldc + ncol] = f2bf(val * oscale);
            } else if (EP == 1) {
                float* C = (float*)Cout + (long)z1 * sCz1 + (long)z2 * sCz2;
                C[(long)r * ldc + ncol] = val;
            } else if (EP == 2) {
                unsigned short* C = (unsigned short*)Cout + (long)z1 * sCz1 + (long)z2 * sCz2;
                C[(long)ncol * ldc + r] = f2bf(val);
            } else if (EP == 3) {
                const float* Rb = R + (long)z1 * sRz1;
                float x = val + Rb[(long)r * ldc + ncol];
                float* C = (float*)Cout + (long)z1 * sCz1;
                C[(long)r * ldc + ncol] = x;
                unsigned short* Cb = C2 + (long)z1 * sC2z1;
                Cb[(long)r * ldc + ncol] = f2bf(x);
            } else { // EP == 4 : exact GELU
                float g = 0.5f * val * (1.0f + erff(val * 0.70710678118654752f));
                unsigned short* C = (unsigned short*)Cout + (long)z1 * sCz1;
                C[(long)r * ldc + ncol] = f2bf(g);
            }
        }
    }
}

// ---------------- Fused multi-camera attention ----------------
// grid: (157 q-tiles, 8 heads, 2 batch), block 256 (8 waves)
// qf : (B,NC,QP,DIM) bf16 (pre-scaled by 1/sqrt(dh)) ; kf : (B,NC,KP,DIM) bf16
// vfT : (B,NC,DIM,KP) bf16 ; attn_out : (B,QP,ND) f32
__global__ __launch_bounds__(256)
void attn_kernel(const unsigned short* __restrict__ qf,
                 const unsigned short* __restrict__ kf,
                 const unsigned short* __restrict__ vfT,
                 float* __restrict__ attn_out) {
    extern __shared__ float smem[];           // [16][SROW] scores/probs
    float* aux  = smem + 16 * SROW;           // [16][16] partial max
    float* aux2 = aux + 256;                  // [16][16] partial sum
    float* auxR = aux2 + 256;                 // [16] 1/rowsum

    const int qtile = blockIdx.x;
    const int m     = blockIdx.y;
    const int b     = blockIdx.z;
    const int wave = threadIdx.x >> 5;
    const int lane = threadIdx.x & 31;
    const int lo16 = lane & 15;
    const int hi   = lane >> 4;

    const int qrow = qtile * 16 + lo16;

    // ---- Phase 1: scores into LDS (scale pre-folded into qf) ----
    // exactly 5 key tiles per wave per cam: fully unrolled, last peeled
#pragma unroll
    for (int cam = 0; cam < NC; ++cam) {
        FragU qa;
        const unsigned short* p =
            qf + (((long)(b * NC + cam) * QP + qrow) * DIM) + m * DH + (hi ? 8 : 0);
        qa.u[0] = *(const uint4*)(p);
        qa.u[1] = *(const uint4*)(p + 16);

        const unsigned short* kbase =
            kf + ((long)(b * NC + cam) * KP) * DIM + m * DH + (hi ? 8 : 0);

        FragU fb;
        {
            const unsigned short* pb = kbase + (long)(wave * 16 + lo16) * DIM;
            fb.u[0] = *(const uint4*)(pb);
            fb.u[1] = *(const uint4*)(pb + 16);
        }
#pragma unroll
        for (int i = 0; i < 4; ++i) {
            int t = wave + 8 * i;
            FragU nf;
            const unsigned short* pn = kbase + (long)((t + 8) * 16 + lo16) * DIM;
            nf.u[0] = *(const uint4*)(pn);
            nf.u[1] = *(const uint4*)(pn + 16);
            v8f s = zero8();
            s = __builtin_amdgcn_wmma_f32_16x16x32_bf16(false, qa.v, false, fb.v,
                                                        (short)0, s, false, false);
            int keyG = cam * KP + t * 16 + lo16;
#pragma unroll
            for (int e = 0; e < 8; ++e)
                smem[(e + 8 * hi) * SROW + keyG] = s[e];
            fb = nf;
        }
        {   // peeled last tile (t = wave + 32)
            v8f s = zero8();
            s = __builtin_amdgcn_wmma_f32_16x16x32_bf16(false, qa.v, false, fb.v,
                                                        (short)0, s, false, false);
            int keyG = cam * KP + (wave + 32) * 16 + lo16;
#pragma unroll
            for (int e = 0; e < 8; ++e)
                smem[(e + 8 * hi) * SROW + keyG] = s[e];
        }
    }
    __syncthreads();

    // ---- Softmax over 3750 valid keys (mask per-cam pads [625,640)) ----
    {
        int r   = threadIdx.x >> 4;     // 0..15 query row
        int seg = threadIdx.x & 15;
        float mx = -1e30f;
        for (int k = seg; k < NC * KP; k += 16)
            if ((k % KP) < K0) mx = fmaxf(mx, smem[r * SROW + k]);
        aux[r * 16 + seg] = mx;
        __syncthreads();
        float rmax = -1e30f;
#pragma unroll
        for (int i = 0; i < 16; ++i) rmax = fmaxf(rmax, aux[r * 16 + i]);
        float ssum = 0.0f;
        for (int k = seg; k < NC * KP; k += 16) {
            float v = 0.0f;
            if ((k % KP) < K0) v = expf(smem[r * SROW + k] - rmax);
            smem[r * SROW + k] = v;
            ssum += v;
        }
        aux2[r * 16 + seg] = ssum;
        __syncthreads();
        float rs = 0.0f;
#pragma unroll
        for (int i = 0; i < 16; ++i) rs += aux2[r * 16 + i];
        if (seg == 0) auxR[r] = 1.0f / rs;
    }
    __syncthreads();

    // ---- Phase 2: out = attn @ vh per (cam, 16-col tile of dh) ----
    for (int u = wave; u < NC * 2; u += 8) {
        int cam = u >> 1;
        int nt  = u & 1;
        int ccol = m * DH + nt * 16 + lo16;     // channel within cam block
        const unsigned short* vb = vfT + (((long)(b * NC + cam) * DIM + ccol) * KP);
        v8f acc = zero8();
        FragU fv;
        {
            const unsigned short* pv = vb + (hi ? 8 : 0);
            fv.u[0] = *(const uint4*)(pv);
            fv.u[1] = *(const uint4*)(pv + 16);
        }
        const float* arow = smem + (long)lo16 * SROW + cam * KP + (hi ? 8 : 0);
        for (int ks = 0; ks < KP / 32 - 1; ++ks) {      // steady state, peel last
            FragU nv;
            const unsigned short* pn = vb + (ks + 1) * 32 + (hi ? 8 : 0);
            nv.u[0] = *(const uint4*)(pn);
            nv.u[1] = *(const uint4*)(pn + 16);
            const float4* ap = (const float4*)(arow + ks * 32);
            float4 c0 = ap[0], c1 = ap[1], c2 = ap[4], c3 = ap[5];
            FragU fa;
            fa.s[0]  = f2bf(c0.x); fa.s[1]  = f2bf(c0.y);
            fa.s[2]  = f2bf(c0.z); fa.s[3]  = f2bf(c0.w);
            fa.s[4]  = f2bf(c1.x); fa.s[5]  = f2bf(c1.y);
            fa.s[6]  = f2bf(c1.z); fa.s[7]  = f2bf(c1.w);
            fa.s[8]  = f2bf(c2.x); fa.s[9]  = f2bf(c2.y);
            fa.s[10] = f2bf(c2.z); fa.s[11] = f2bf(c2.w);
            fa.s[12] = f2bf(c3.x); fa.s[13] = f2bf(c3.y);
            fa.s[14] = f2bf(c3.z); fa.s[15] = f2bf(c3.w);
            acc = __builtin_amdgcn_wmma_f32_16x16x32_bf16(false, fa.v, false, fv.v,
                                                          (short)0, acc, false, false);
            fv = nv;
        }
        {   // peeled last K-step (ks = 19)
            const float4* ap = (const float4*)(arow + (KP / 32 - 1) * 32);
            float4 c0 = ap[0], c1 = ap[1], c2 = ap[4], c3 = ap[5];
            FragU fa;
            fa.s[0]  = f2bf(c0.x); fa.s[1]  = f2bf(c0.y);
            fa.s[2]  = f2bf(c0.z); fa.s[3]  = f2bf(c0.w);
            fa.s[4]  = f2bf(c1.x); fa.s[5]  = f2bf(c1.y);
            fa.s[6]  = f2bf(c1.z); fa.s[7]  = f2bf(c1.w);
            fa.s[8]  = f2bf(c2.x); fa.s[9]  = f2bf(c2.y);
            fa.s[10] = f2bf(c2.z); fa.s[11] = f2bf(c2.w);
            fa.s[12] = f2bf(c3.x); fa.s[13] = f2bf(c3.y);
            fa.s[14] = f2bf(c3.z); fa.s[15] = f2bf(c3.w);
            acc = __builtin_amdgcn_wmma_f32_16x16x32_bf16(false, fa.v, false, fv.v,
                                                          (short)0, acc, false, false);
        }
#pragma unroll
        for (int e = 0; e < 8; ++e) {
            int rrow = e + 8 * hi;
            int qg = qtile * 16 + rrow;
            attn_out[((long)b * QP + qg) * ND + cam * DIM + ccol] = acc[e] * auxR[rrow];
        }
    }
}

// ---------------- LayerNorm over 1536, write normalized bf16 ----------------
__global__ void ln1536(const float* __restrict__ X, const float* __restrict__ g,
                       const float* __restrict__ bt, unsigned short* __restrict__ Y) {
    __shared__ float red[256];
    long row = blockIdx.x;
    int tid = threadIdx.x;
    const float* xr = X + row * ND;
    float s = 0.0f;
    for (int c = tid; c < ND; c += 256) s += xr[c];
    red[tid] = s; __syncthreads();
    for (int off = 128; off > 0; off >>= 1) {
        if (tid < off) red[tid] += red[tid + off];
        __syncthreads();
    }
    float mean = red[0] * (1.0f / ND);
    __syncthreads();
    float v = 0.0f;
    for (int c = tid; c < ND; c += 256) { float d = xr[c] - mean; v += d * d; }
    red[tid] = v; __syncthreads();
    for (int off = 128; off > 0; off >>= 1) {
        if (tid < off) red[tid] += red[tid + off];
        __syncthreads();
    }
    float is = rsqrtf(red[0] * (1.0f / ND) + 1e-5f);
    unsigned short* yr = Y + row * ND;
    for (int c = tid; c < ND; c += 256)
        yr[c] = f2bf((xr[c] - mean) * is * g[c] + bt[c]);
}

// ---------------- Final: LN(h2)*g+b + x, transpose to (B,D,H,W) ----------------
__global__ void final_ln_out(const float* __restrict__ h2, const float* __restrict__ x,
                             const float* __restrict__ g, const float* __restrict__ bt,
                             float* __restrict__ out) {
    __shared__ float red[256];
    int q = blockIdx.x;     // 0..2499
    int b = blockIdx.y;
    int c = threadIdx.x;
    long row = (long)b * QP + q;
    float val = h2[row * DIM + c];
    red[c] = val; __syncthreads();
    for (int off = 128; off > 0; off >>= 1) {
        if (c < off) red[c] += red[c + off];
        __syncthreads();
    }
    float mean = red[0] * (1.0f / DIM);
    __syncthreads();
    float d = val - mean;
    red[c] = d * d; __syncthreads();
    for (int off = 128; off > 0; off >>= 1) {
        if (c < off) red[c] += red[c + off];
        __syncthreads();
    }
    float is = rsqrtf(red[0] * (1.0f / DIM) + 1e-5f);
    float y = d * is * g[c] + bt[c] + x[row * DIM + c];
    out[((long)b * DIM + c) * QLEN + q] = y;
}

// ---------------- Launch ----------------
extern "C" void kernel_launch(void* const* d_in, const int* in_sizes, int n_in,
                              void* d_out, int out_size, void* d_ws, size_t ws_size,
                              hipStream_t stream) {
    const float* q_in   = (const float*)d_in[0];
    const float* k_in   = (const float*)d_in[1];
    const float* v_in   = (const float*)d_in[2];
    const float* Wq     = (const float*)d_in[3];
    const float* bq     = (const float*)d_in[4];
    const float* Wk     = (const float*)d_in[5];
    const float* bk     = (const float*)d_in[6];
    const float* Wv     = (const float*)d_in[7];
    const float* bv     = (const float*)d_in[8];
    const float* Wproj  = (const float*)d_in[9];
    const float* bproj  = (const float*)d_in[10];
    const float* Waddq  = (const float*)d_in[11];
    const float* baddq  = (const float*)d_in[12];
    const float* W1     = (const float*)d_in[13];
    const float* b1     = (const float*)d_in[14];
    const float* W2     = (const float*)d_in[15];
    const float* b2     = (const float*)d_in[16];
    const float* g_pre  = (const float*)d_in[17];
    const float* b_pre  = (const float*)d_in[18];
    const float* g_norm = (const float*)d_in[19];
    const float* b_norm = (const float*)d_in[20];
    float* out = (float*)d_out;

    char* ws = (char*)d_ws;
    size_t cur = 0;
    auto alloc = [&](size_t bytes) -> char* {
        char* p = ws + cur;
        cur = (cur + bytes + 255) & ~(size_t)255;
        return p;
    };

    unsigned short* WqT    = (unsigned short*)alloc((size_t)DIM * DIM * 2);
    unsigned short* WkT    = (unsigned short*)alloc((size_t)DIM * DIM * 2);
    unsigned short* WvT    = (unsigned short*)alloc((size_t)DIM * DIM * 2);
    unsigned short* WaddqT = (unsigned short*)alloc((size_t)DIM * ND * 2);
    unsigned short* WprojT = (unsigned short*)alloc((size_t)DIM * ND * 2);
    unsigned short* W1T    = (unsigned short*)alloc((size_t)512 * DIM * 2);
    unsigned short* W2T    = (unsigned short*)alloc((size_t)DIM * 512 * 2);
    unsigned short* Xq     = (unsigned short*)alloc((size_t)BATCH * QP * ND * 2);
    unsigned short* Xk     = (unsigned short*)alloc((size_t)BATCH * NC * KP * DIM * 2);
    unsigned short* Xv     = (unsigned short*)alloc((size_t)BATCH * NC * KP * DIM * 2);
    unsigned short* qfB    = (unsigned short*)alloc((size_t)BATCH * NC * QP * DIM * 2);
    unsigned short* kfB    = (unsigned short*)alloc((size_t)BATCH * NC * KP * DIM * 2);
    unsigned short* vfT    = (unsigned short*)alloc((size_t)BATCH * NC * DIM * KP * 2);
    float*          addq   = (float*)alloc((size_t)BATCH * QP * DIM * 4);
    float*          attnO  = (float*)alloc((size_t)BATCH * QP * ND * 4);
    unsigned short* xa_n   = (unsigned short*)alloc((size_t)BATCH * QP * ND * 2);
    float*          xbuf   = (float*)alloc((size_t)BATCH * QP * DIM * 4);
    unsigned short* x_bf   = (unsigned short*)alloc((size_t)BATCH * QP * DIM * 2);
    unsigned short* h1_bf  = (unsigned short*)alloc((size_t)BATCH * QP * 512 * 2);
    float*          h2     = (float*)alloc((size_t)BATCH * QP * DIM * 4);

    // 1. weight transpose/convert
    auto cvt = [&](const float* W, unsigned short* Wt, int K, int N) {
        long tot = (long)K * N;
        cvt_w_t<<<dim3((tot + 255) / 256), dim3(256), 0, stream>>>(W, Wt, K, N);
    };
    cvt(Wq, WqT, DIM, DIM);
    cvt(Wk, WkT, DIM, DIM);
    cvt(Wv, WvT, DIM, DIM);
    cvt(Waddq, WaddqT, ND, DIM);
    cvt(Wproj, WprojT, ND, DIM);
    cvt(W1, W1T, DIM, 512);
    cvt(W2, W2T, 512, DIM);

    // 2. activation packing (+2x2 avg pool for k,v)
    {
        long tot = (long)BATCH * QP * ND;
        pack_q<<<dim3((tot + 255) / 256), dim3(256), 0, stream>>>(q_in, Xq);
        long tp = (long)BATCH * NC * KP * DIM;
        pack_pool<<<dim3((tp + 255) / 256), dim3(256), 0, stream>>>(k_in, Xk);
        pack_pool<<<dim3((tp + 255) / 256), dim3(256), 0, stream>>>(v_in, Xv);
    }

    const float attn_scale = 0.17677669529663687f;   // 1/sqrt(DH), folded into qf

    // 3. qf = (Xq(per-cam slice) @ Wq + bq) * scale  -> bf16 (B,NC,QP,DIM)
    gemm_bf16<0><<<dim3(DIM / 64, (QP + 63) / 64, BATCH * NC), dim3(256), 0, stream>>>(
        Xq, (long)QP * ND, (long)DIM, ND, WqT, bq,
        qfB, (long)NC * QP * DIM, (long)QP * DIM, DIM,
        nullptr, 0, nullptr, 0, QP, DIM, DIM, NC, attn_scale);

    // 4. kf = Xk @ Wk + bk  -> bf16 (B,NC,KP,DIM)
    gemm_bf16<0><<<dim3(DIM / 64, KP / 64, BATCH * NC), dim3(256), 0, stream>>>(
        Xk, (long)NC * KP * DIM, (long)KP * DIM, DIM, WkT, bk,
        kfB, (long)NC * KP * DIM, (long)KP * DIM, DIM,
        nullptr, 0, nullptr, 0, KP, DIM, DIM, NC, 1.0f);

    // 5. vfT = (Xv @ Wv + bv)^T  -> bf16 (B,NC,DIM,KP)
    gemm_bf16<2><<<dim3(DIM / 64, KP / 64, BATCH * NC), dim3(256), 0, stream>>>(
        Xv, (long)NC * KP * DIM, (long)KP * DIM, DIM, WvT, bv,
        vfT, (long)NC * DIM * KP, (long)DIM * KP, KP,
        nullptr, 0, nullptr, 0, KP, DIM, DIM, NC, 1.0f);

    // 6. add_q = Xq @ Waddq + baddq -> f32 (B,QP,DIM)
    gemm_bf16<1><<<dim3(DIM / 64, (QP + 63) / 64, BATCH), dim3(256), 0, stream>>>(
        Xq, (long)QP * ND, 0, ND, WaddqT, baddq,
        addq, (long)QP * DIM, 0, DIM,
        nullptr, 0, nullptr, 0, QP, DIM, ND, 1, 1.0f);

    // 7. fused attention
    {
        size_t smem = (size_t)(16 * SROW + 256 + 256 + 16) * sizeof(float);
        attn_kernel<<<dim3(QP / 16, NHEAD, BATCH), dim3(256), smem, stream>>>(
            qfB, kfB, vfT, attnO);
    }

    // 8. pre-norm over 1536 -> bf16
    ln1536<<<dim3(BATCH * QP), dim3(256), 0, stream>>>(attnO, g_pre, b_pre, xa_n);

    // 9. x = xa_n @ Wproj + bproj + add_q  -> f32 + bf16
    gemm_bf16<3><<<dim3(DIM / 64, (QP + 63) / 64, BATCH), dim3(256), 0, stream>>>(
        xa_n, (long)QP * ND, 0, ND, WprojT, bproj,
        xbuf, (long)QP * DIM, 0, DIM,
        addq, (long)QP * DIM, x_bf, (long)QP * DIM, QP, DIM, ND, 1, 1.0f);

    // 10. h1 = gelu(x @ W1 + b1) -> bf16 (B,QP,512)
    gemm_bf16<4><<<dim3(512 / 64, (QP + 63) / 64, BATCH), dim3(256), 0, stream>>>(
        x_bf, (long)QP * DIM, 0, DIM, W1T, b1,
        h1_bf, (long)QP * 512, 0, 512,
        nullptr, 0, nullptr, 0, QP, 512, DIM, 1, 1.0f);

    // 11. h2 = h1 @ W2 + b2 -> f32 (B,QP,DIM)
    gemm_bf16<1><<<dim3(DIM / 64, (QP + 63) / 64, BATCH), dim3(256), 0, stream>>>(
        h1_bf, (long)QP * 512, 0, 512, W2T, b2,
        h2, (long)QP * DIM, 0, DIM,
        nullptr, 0, nullptr, 0, QP, DIM, 512, 1, 1.0f);

    // 12. out = transpose(x + LN(h2)*g_norm + b_norm)
    final_ln_out<<<dim3(QLEN, BATCH), dim3(256), 0, stream>>>(h2, xbuf, g_norm, b_norm, out);

    (void)in_sizes; (void)n_in; (void)out_size; (void)ws_size;
}
